// FlashPhiAttention_2121713845079
// MI455X (gfx1250) — matrix operand order repaired and checked by reading, not compiled, not run on hardware
//
#include <hip/hip_runtime.h>
#include <stdint.h>

#ifndef USE_ASYNC_LDS
#define USE_ASYNC_LDS 1   // gfx1250 async global->LDS staging in the GEMM
#endif

// ---------- problem constants ----------
constexpr int Bc   = 2;
constexpr int Sc   = 2048;
constexpr int Dc   = 2560;
constexpr int Hc   = 32;
constexpr int HKVc = 8;
constexpr int HDc  = 80;
constexpr int Gc   = Hc / HKVc;             // 4
constexpr int Tc   = Bc * Sc;               // 4096
constexpr int NQKV = (Hc + 2 * HKVc) * HDc; // 3840
constexpr float SCALE = 0.111803398874989f; // 80^-0.5

// ---------- CDNA5 WMMA types ----------
typedef __attribute__((ext_vector_type(16))) __bf16        v16bf;
typedef __attribute__((ext_vector_type(8)))  float         v8f;
typedef __attribute__((ext_vector_type(4)))  unsigned int  u32x4;

union Frag {
  u32x4 q[2];
  v16bf v;
};

__device__ __forceinline__ v8f wmma_bf16(const Frag& a, const Frag& b, v8f c) {
  // D(16x16 f32) = A(16x32 bf16) x B(32x16 bf16) + C
  return __builtin_amdgcn_wmma_f32_16x16x32_bf16(
      /*neg_a=*/false, a.v, /*neg_b=*/false, b.v,
      /*c_mod=*/(short)0, c, /*reuse_a=*/false, /*reuse_b=*/false);
}

__device__ __forceinline__ unsigned short f2bf(float f) {
  unsigned int u = __float_as_uint(f);
  u += 0x7FFFu + ((u >> 16) & 1u);  // round-to-nearest-even
  return (unsigned short)(u >> 16);
}

#if USE_ASYNC_LDS
// Async DMA: 16 bytes per lane, global -> LDS, tracked by ASYNCcnt.
// LDS flat addresses keep the byte offset in addr[31:0] (ISA 10.2 aperture
// mapping), so truncation yields the LDS address the instruction needs.
__device__ __forceinline__ void async_load_b128(const void* gaddr, void* lds) {
  unsigned loff = (unsigned)(unsigned long long)lds;
  asm volatile("global_load_async_to_lds_b128 %0, %1, off"
               :: "v"(loff), "v"(gaddr)
               : "memory");
}
__device__ __forceinline__ void wait_async() {
  asm volatile("s_wait_asynccnt 0" ::: "memory");
}
#endif

// ---------- fp32 -> bf16 cast ----------
__global__ void k_cast_bf16(const float* __restrict__ src,
                            unsigned short* __restrict__ dst, long long n) {
  long long i = (long long)blockIdx.x * blockDim.x + threadIdx.x;
  if (i < n) dst[i] = f2bf(src[i]);
}

// ---------- GEMM: C[M,N] = A[M,K](bf16) * W[N,K](bf16)^T + bias, fp32 out ----------
// block tile 128x128, K-step 32, 8 waves -> each wave a 64(M)x32(N) tile.
#if USE_ASYNC_LDS
// Double-buffered: async-copy next K-tile to LDS while doing WMMAs on current.
__global__ __launch_bounds__(256)
void k_gemm_bf16(const unsigned short* __restrict__ A,
                 const unsigned short* __restrict__ W,
                 const float* __restrict__ bias,
                 float* __restrict__ C,
                 int M, int N, int K) {
  __shared__ unsigned short As[2][128][40] __attribute__((aligned(16)));
  __shared__ unsigned short Bs[2][128][40] __attribute__((aligned(16)));

  const int tid  = threadIdx.x;
  const int lane = tid & 31;
  const int w    = tid >> 5;
  const int wm   = w & 1;       // 0..1 (M)
  const int wn   = w >> 1;      // 0..3 (N)
  const int lr   = lane & 15;
  const int half = lane >> 4;
  const int m0   = blockIdx.y * 128;
  const int n0   = blockIdx.x * 128;

  v8f acc[4][2];
#pragma unroll
  for (int mi = 0; mi < 4; ++mi)
#pragma unroll
    for (int ni = 0; ni < 2; ++ni)
#pragma unroll
      for (int j = 0; j < 8; ++j) acc[mi][ni][j] = 0.0f;

  const int ldr = tid >> 1;          // 0..127
  const int ldc = (tid & 1) * 16;    // 0 or 16

  const unsigned short* ag = A + (size_t)(m0 + ldr) * K + ldc;
  const unsigned short* wg = W + (size_t)(n0 + ldr) * K + ldc;

  // prologue: stage K-tile 0 into buffer 0
  async_load_b128(ag,     &As[0][ldr][ldc]);
  async_load_b128(ag + 8, &As[0][ldr][ldc + 8]);
  async_load_b128(wg,     &Bs[0][ldr][ldc]);
  async_load_b128(wg + 8, &Bs[0][ldr][ldc + 8]);
  wait_async();
  __syncthreads();

  int buf = 0;
  for (int k0 = 0; k0 < K; k0 += 32) {
    const int nxt = buf ^ 1;
    if (k0 + 32 < K) {  // prefetch next K-tile while computing
      const unsigned short* ag2 = ag + k0 + 32;
      const unsigned short* wg2 = wg + k0 + 32;
      async_load_b128(ag2,     &As[nxt][ldr][ldc]);
      async_load_b128(ag2 + 8, &As[nxt][ldr][ldc + 8]);
      async_load_b128(wg2,     &Bs[nxt][ldr][ldc]);
      async_load_b128(wg2 + 8, &Bs[nxt][ldr][ldc + 8]);
    }

    Frag af[4], bf2[2];
#pragma unroll
    for (int mi = 0; mi < 4; ++mi) {
      int r = wm * 64 + mi * 16 + lr;
      af[mi].q[0] = *(const u32x4*)&As[buf][r][half * 8];
      af[mi].q[1] = *(const u32x4*)&As[buf][r][16 + half * 8];
    }
#pragma unroll
    for (int ni = 0; ni < 2; ++ni) {
      int c2 = wn * 32 + ni * 16 + lr;
      bf2[ni].q[0] = *(const u32x4*)&Bs[buf][c2][half * 8];
      bf2[ni].q[1] = *(const u32x4*)&Bs[buf][c2][16 + half * 8];
    }
#pragma unroll
    for (int mi = 0; mi < 4; ++mi)
#pragma unroll
      for (int ni = 0; ni < 2; ++ni)
        acc[mi][ni] = wmma_bf16(af[mi], bf2[ni], acc[mi][ni]);

    wait_async();      // next tile landed (had the whole compute to finish)
    __syncthreads();   // everyone done reading `buf`; safe to overwrite next iter
    buf = nxt;
  }

#pragma unroll
  for (int mi = 0; mi < 4; ++mi)
#pragma unroll
    for (int ni = 0; ni < 2; ++ni) {
      int row0 = m0 + wm * 64 + mi * 16 + 8 * half;
      int col  = n0 + wn * 32 + ni * 16 + lr;
      float bv = bias[col];
#pragma unroll
      for (int r = 0; r < 8; ++r)
        C[(size_t)(row0 + r) * N + col] = acc[mi][ni][r] + bv;
    }
}
#else
// Fallback: global->VGPR->LDS staging.
__global__ __launch_bounds__(256)
void k_gemm_bf16(const unsigned short* __restrict__ A,
                 const unsigned short* __restrict__ W,
                 const float* __restrict__ bias,
                 float* __restrict__ C,
                 int M, int N, int K) {
  __shared__ unsigned short As[128][40] __attribute__((aligned(16)));
  __shared__ unsigned short Bs[128][40] __attribute__((aligned(16)));

  const int tid  = threadIdx.x;
  const int lane = tid & 31;
  const int w    = tid >> 5;
  const int wm   = w & 1;
  const int wn   = w >> 1;
  const int lr   = lane & 15;
  const int half = lane >> 4;
  const int m0   = blockIdx.y * 128;
  const int n0   = blockIdx.x * 128;

  v8f acc[4][2];
#pragma unroll
  for (int mi = 0; mi < 4; ++mi)
#pragma unroll
    for (int ni = 0; ni < 2; ++ni)
#pragma unroll
      for (int j = 0; j < 8; ++j) acc[mi][ni][j] = 0.0f;

  const int ldr = tid >> 1;
  const int ldc = (tid & 1) * 16;

  for (int k0 = 0; k0 < K; k0 += 32) {
    const unsigned short* ag = A + (size_t)(m0 + ldr) * K + k0 + ldc;
    const unsigned short* wg = W + (size_t)(n0 + ldr) * K + k0 + ldc;
    u32x4 a0 = *(const u32x4*)ag;
    u32x4 a1 = *(const u32x4*)(ag + 8);
    u32x4 b0 = *(const u32x4*)wg;
    u32x4 b1 = *(const u32x4*)(wg + 8);
    __syncthreads();
    *(u32x4*)&As[ldr][ldc]     = a0;
    *(u32x4*)&As[ldr][ldc + 8] = a1;
    *(u32x4*)&Bs[ldr][ldc]     = b0;
    *(u32x4*)&Bs[ldr][ldc + 8] = b1;
    __syncthreads();

    Frag af[4], bf2[2];
#pragma unroll
    for (int mi = 0; mi < 4; ++mi) {
      int r = wm * 64 + mi * 16 + lr;
      af[mi].q[0] = *(const u32x4*)&As[r][half * 8];
      af[mi].q[1] = *(const u32x4*)&As[r][16 + half * 8];
    }
#pragma unroll
    for (int ni = 0; ni < 2; ++ni) {
      int c2 = wn * 32 + ni * 16 + lr;
      bf2[ni].q[0] = *(const u32x4*)&Bs[c2][half * 16];
      bf2[ni].q[1] = *(const u32x4*)&Bs[c2][half * 16 + 8];
    }
#pragma unroll
    for (int mi = 0; mi < 4; ++mi)
#pragma unroll
      for (int ni = 0; ni < 2; ++ni)
        acc[mi][ni] = wmma_bf16(af[mi], bf2[ni], acc[mi][ni]);
  }

#pragma unroll
  for (int mi = 0; mi < 4; ++mi)
#pragma unroll
    for (int ni = 0; ni < 2; ++ni) {
      int row0 = m0 + wm * 64 + mi * 16 + 8 * half;
      int col  = n0 + wn * 32 + ni * 16 + lr;
      float bv = bias[col];
#pragma unroll
      for (int r = 0; r < 8; ++r)
        C[(size_t)(row0 + r) * N + col] = acc[mi][ni][r] + bv;
    }
}
#endif  // USE_ASYNC_LDS

// ---------- RoPE for Q (scale folded in), layout -> [B*H, S, 80] bf16 ----------
__global__ void k_rope_q(const float* __restrict__ qkv,
                         const float* __restrict__ cosT,
                         const float* __restrict__ sinT,
                         unsigned short* __restrict__ Q) {
  long long i = (long long)blockIdx.x * blockDim.x + threadIdx.x;
  if (i >= (long long)Tc * Hc * 40) return;
  int t = (int)(i / (Hc * 40));
  int rem = (int)(i % (Hc * 40));
  int h = rem / 40, d = rem % 40;
  const float* row = qkv + (size_t)t * NQKV + h * HDc;
  float x1 = row[d], x2 = row[40 + d];
  float c = cosT[(size_t)t * 40 + d], s = sinT[(size_t)t * 40 + d];
  int b = t / Sc, sp = t % Sc;
  size_t base = ((size_t)(b * Hc + h) * Sc + sp) * HDc;
  Q[base + d]      = f2bf((x1 * c - x2 * s) * SCALE);
  Q[base + 40 + d] = f2bf((x1 * s + x2 * c) * SCALE);
}

// ---------- RoPE for K, layout -> [B*HKV, S, 80] bf16 ----------
__global__ void k_rope_k(const float* __restrict__ qkv,
                         const float* __restrict__ cosT,
                         const float* __restrict__ sinT,
                         unsigned short* __restrict__ Kd) {
  long long i = (long long)blockIdx.x * blockDim.x + threadIdx.x;
  if (i >= (long long)Tc * HKVc * 40) return;
  int t = (int)(i / (HKVc * 40));
  int rem = (int)(i % (HKVc * 40));
  int h = rem / 40, d = rem % 40;
  const float* row = qkv + (size_t)t * NQKV + Hc * HDc + h * HDc;
  float x1 = row[d], x2 = row[40 + d];
  float c = cosT[(size_t)t * 40 + d], s = sinT[(size_t)t * 40 + d];
  int b = t / Sc, sp = t % Sc;
  size_t base = ((size_t)(b * HKVc + h) * Sc + sp) * HDc;
  Kd[base + d]      = f2bf(x1 * c - x2 * s);
  Kd[base + 40 + d] = f2bf(x1 * s + x2 * c);
}

// ---------- V reshape/convert -> [B*HKV, S, 80] bf16 ----------
__global__ void k_copy_v(const float* __restrict__ qkv,
                         unsigned short* __restrict__ Vd) {
  long long i = (long long)blockIdx.x * blockDim.x + threadIdx.x;
  if (i >= (long long)Tc * HKVc * HDc) return;
  int t = (int)(i / (HKVc * HDc));
  int rem = (int)(i % (HKVc * HDc));
  int h = rem / HDc, d = rem % HDc;
  float v = qkv[(size_t)t * NQKV + (Hc + HKVc) * HDc + h * HDc + d];
  int b = t / Sc, sp = t % Sc;
  Vd[((size_t)(b * HKVc + h) * Sc + sp) * HDc + d] = f2bf(v);
}

// ---------- flash attention (causal, GQA), bf16 in/out ----------
// grid: (S/128, B*H); block 256 (8 waves). Wave w owns q rows [16w, 16w+16).
__global__ __launch_bounds__(256)
void k_flash(const unsigned short* __restrict__ Q,
             const unsigned short* __restrict__ Kd,
             const unsigned short* __restrict__ Vd,
             unsigned short* __restrict__ O) {
  __shared__ unsigned short Qs[128][96] __attribute__((aligned(16))); // K padded 80->96
  __shared__ unsigned short Ks[64][96]  __attribute__((aligned(16)));
  __shared__ unsigned short VTs[80][64] __attribute__((aligned(16))); // V transposed
  __shared__ unsigned short Ps[8][16][64] __attribute__((aligned(16))); // per-wave P

  const int tid  = threadIdx.x;
  const int lane = tid & 31;
  const int w    = tid >> 5;
  const int lr   = lane & 15;
  const int half = lane >> 4;

  const int qb = blockIdx.x * 128;
  const int bh = blockIdx.y;
  const int b  = bh / Hc;
  const int h  = bh % Hc;
  const int hk = h / Gc;

  const unsigned short* Qg = Q  + ((size_t)bh * Sc + qb) * HDc;
  const unsigned short* Kg = Kd + ((size_t)(b * HKVc + hk) * Sc) * HDc;
  const unsigned short* Vg = Vd + ((size_t)(b * HKVc + hk) * Sc) * HDc;

  for (int i = tid; i < 128 * 96; i += 256) {
    int r = i / 96, c = i % 96;
    Qs[r][c] = (c < HDc) ? Qg[(size_t)r * HDc + c] : (unsigned short)0;
  }

  float m_i[8], l_i[8];
  v8f acc_o[5];
#pragma unroll
  for (int r = 0; r < 8; ++r) { m_i[r] = -3.0e38f; l_i[r] = 0.0f; }
#pragma unroll
  for (int nt = 0; nt < 5; ++nt)
#pragma unroll
    for (int j = 0; j < 8; ++j) acc_o[nt][j] = 0.0f;

  const int nkb = qb / 64 + 2; // causal: keys up to qb+127
  for (int kb = 0; kb < nkb; ++kb) {
    const int kbase = kb * 64;
    __syncthreads();
    for (int i = tid; i < 64 * 96; i += 256) {
      int r = i / 96, c = i % 96;
      Ks[r][c] = (c < HDc) ? Kg[(size_t)(kbase + r) * HDc + c] : (unsigned short)0;
    }
    for (int i = tid; i < 80 * 64; i += 256) {
      int d = i / 64, kx = i % 64;
      VTs[d][kx] = Vg[(size_t)(kbase + kx) * HDc + d];
    }
    __syncthreads();

    if (kbase <= qb + w * 16 + 15) {  // wave-uniform causal skip
      v8f sc[4];
#pragma unroll
      for (int nt = 0; nt < 4; ++nt)
#pragma unroll
        for (int j = 0; j < 8; ++j) sc[nt][j] = 0.0f;

      // scores = Q (16x96) * K^T (96x64)
#pragma unroll
      for (int kk = 0; kk < 96; kk += 32) {
        Frag a;
        a.q[0] = *(const u32x4*)&Qs[w * 16 + lr][kk + half * 8];
        a.q[1] = *(const u32x4*)&Qs[w * 16 + lr][kk + 16 + half * 8];
#pragma unroll
        for (int nt = 0; nt < 4; ++nt) {
          Frag bb;
          bb.q[0] = *(const u32x4*)&Ks[nt * 16 + lr][kk + half * 8];
          bb.q[1] = *(const u32x4*)&Ks[nt * 16 + lr][kk + 16 + half * 8];
          sc[nt] = wmma_bf16(a, bb, sc[nt]);
        }
      }

      // causal mask + online softmax
#pragma unroll
      for (int r = 0; r < 8; ++r) {
        int qg = qb + w * 16 + r + 8 * half;
#pragma unroll
        for (int nt = 0; nt < 4; ++nt) {
          int kg = kbase + nt * 16 + lr;
          if (kg > qg) sc[nt][r] = -3.0e38f;
        }
        float rm = -3.0e38f;
#pragma unroll
        for (int nt = 0; nt < 4; ++nt) rm = fmaxf(rm, sc[nt][r]);
#pragma unroll
        for (int mk = 8; mk >= 1; mk >>= 1)
          rm = fmaxf(rm, __shfl_xor(rm, mk, 32));
        float nm = fmaxf(m_i[r], rm);
        float al = __expf(m_i[r] - nm);
        float rs = 0.0f;
#pragma unroll
        for (int nt = 0; nt < 4; ++nt) {
          float p = __expf(sc[nt][r] - nm);
          rs += p;
          Ps[w][r + 8 * half][nt * 16 + lr] = f2bf(p);
        }
#pragma unroll
        for (int mk = 8; mk >= 1; mk >>= 1)
          rs += __shfl_xor(rs, mk, 32);
        l_i[r] = l_i[r] * al + rs;
        m_i[r] = nm;
#pragma unroll
        for (int nt = 0; nt < 5; ++nt) acc_o[nt][r] *= al;
      }

      // O += P (16x64) * V (64x80)
#pragma unroll
      for (int k2 = 0; k2 < 64; k2 += 32) {
        Frag a;
        a.q[0] = *(const u32x4*)&Ps[w][lr][k2 + half * 8];
        a.q[1] = *(const u32x4*)&Ps[w][lr][k2 + 16 + half * 8];
#pragma unroll
        for (int nt = 0; nt < 5; ++nt) {
          Frag bb;
          bb.q[0] = *(const u32x4*)&VTs[nt * 16 + lr][k2 + half * 8];
          bb.q[1] = *(const u32x4*)&VTs[nt * 16 + lr][k2 + 16 + half * 8];
          acc_o[nt] = wmma_bf16(a, bb, acc_o[nt]);
        }
      }
    }
  }

  // write O[t][h*80 + d] bf16
#pragma unroll
  for (int r = 0; r < 8; ++r) {
    float inv = 1.0f / l_i[r];
    long long t = (long long)b * Sc + qb + w * 16 + r + 8 * half;
#pragma unroll
    for (int nt = 0; nt < 5; ++nt)
      O[(size_t)t * (Hc * HDc) + h * HDc + nt * 16 + lr] =
          f2bf(acc_o[nt][r] * inv);
  }
}

// ---------- host-side launch ----------
extern "C" void kernel_launch(void* const* d_in, const int* in_sizes, int n_in,
                              void* d_out, int out_size, void* d_ws, size_t ws_size,
                              hipStream_t stream) {
  const float* hidden = (const float*)d_in[0];
  const float* wqkv_w = (const float*)d_in[1];
  const float* wqkv_b = (const float*)d_in[2];
  const float* out_w  = (const float*)d_in[3];
  const float* out_b  = (const float*)d_in[4];
  const float* cosT   = (const float*)d_in[5];
  const float* sinT   = (const float*)d_in[6];
  float* out = (float*)d_out;

  char* ws = (char*)d_ws;
  size_t off = 0;
  auto wsAlloc = [&](size_t bytes) -> char* {
    char* p = ws + off;
    off += (bytes + 255) & ~(size_t)255;
    return p;
  };

  unsigned short* hid_bf = (unsigned short*)wsAlloc((size_t)Tc * Dc * 2);
  unsigned short* wq_bf  = (unsigned short*)wsAlloc((size_t)NQKV * Dc * 2);
  unsigned short* ow_bf  = (unsigned short*)wsAlloc((size_t)Dc * (Hc * HDc) * 2);
  float*          qkv    = (float*)wsAlloc((size_t)Tc * NQKV * 4);
  unsigned short* qrot   = (unsigned short*)wsAlloc((size_t)Tc * Hc * HDc * 2);
  unsigned short* krot   = (unsigned short*)wsAlloc((size_t)Tc * HKVc * HDc * 2);
  unsigned short* vrot   = (unsigned short*)wsAlloc((size_t)Tc * HKVc * HDc * 2);
  unsigned short* attn   = (unsigned short*)wsAlloc((size_t)Tc * Hc * HDc * 2);
  (void)ws_size; (void)in_sizes; (void)n_in; (void)out_size;

  // 1) casts to bf16
  {
    long long n = (long long)Tc * Dc;
    k_cast_bf16<<<(unsigned)((n + 255) / 256), 256, 0, stream>>>(hidden, hid_bf, n);
    n = (long long)NQKV * Dc;
    k_cast_bf16<<<(unsigned)((n + 255) / 256), 256, 0, stream>>>(wqkv_w, wq_bf, n);
    n = (long long)Dc * Hc * HDc;
    k_cast_bf16<<<(unsigned)((n + 255) / 256), 256, 0, stream>>>(out_w, ow_bf, n);
  }

  // 2) QKV GEMM: [T,3840] = hid[T,2560] x wqkv^T + b
  {
    dim3 grid(NQKV / 128, Tc / 128);
    k_gemm_bf16<<<grid, 256, 0, stream>>>(hid_bf, wq_bf, wqkv_b, qkv, Tc, NQKV, Dc);
  }

  // 3) RoPE + reshape
  {
    long long n = (long long)Tc * Hc * 40;
    k_rope_q<<<(unsigned)((n + 255) / 256), 256, 0, stream>>>(qkv, cosT, sinT, qrot);
    n = (long long)Tc * HKVc * 40;
    k_rope_k<<<(unsigned)((n + 255) / 256), 256, 0, stream>>>(qkv, cosT, sinT, krot);
    n = (long long)Tc * HKVc * HDc;
    k_copy_v<<<(unsigned)((n + 255) / 256), 256, 0, stream>>>(qkv, vrot);
  }

  // 4) flash attention
  {
    dim3 grid(Sc / 128, Bc * Hc);
    k_flash<<<grid, 256, 0, stream>>>(qrot, krot, vrot, attn);
  }

  // 5) output projection: out[T,2560] = attn[T,2560] x out_w^T + out_b
  {
    dim3 grid(Dc / 128, Tc / 128);
    k_gemm_bf16<<<grid, 256, 0, stream>>>(attn, ow_bf, out_b, out, Tc, Dc, Dc);
  }
}
